// WindowAttention_11244224381555
// MI455X (gfx1250) — matrix operand order repaired and checked by reading, not compiled, not run on hardware
//
#include <hip/hip_runtime.h>
#include <hip/hip_bf16.h>

// ---------- types ----------
typedef __bf16 bf16;
typedef bf16  v16bf __attribute__((ext_vector_type(16)));
typedef float v8f   __attribute__((ext_vector_type(8)));

union Frag32 {
    uint4 q[2];
    v16bf v;
};

static __device__ inline unsigned short f2bf(float f) {
    // round-to-nearest-even fp32 -> bf16
    unsigned u = __builtin_bit_cast(unsigned, f);
    unsigned r = u + 0x7FFFu + ((u >> 16) & 1u);
    return (unsigned short)(r >> 16);
}

static __device__ inline v16bf load_frag(const unsigned short* p0, const unsigned short* p1) {
    Frag32 u;
    u.q[0] = *reinterpret_cast<const uint4*>(p0);
    u.q[1] = *reinterpret_cast<const uint4*>(p1);
    return u.v;
}

static __device__ inline v8f wmma_bf16(v16bf a, v16bf b, v8f c) {
    // D = A(16x32 bf16) x B(32x16 bf16) + C(16x16 f32)
    return __builtin_amdgcn_wmma_f32_16x16x32_bf16(false, a, false, b, (short)0, c, false, false);
}

// ---------- problem constants ----------
constexpr int BWIN = 2048, NTOK = 49, CDIM = 512, HEADS = 16, HD = 32, COUT3 = 1536;
constexpr int MROWS = BWIN * NTOK;               // 100352 tokens, /16 = 6272 exactly
constexpr float SCALE = 0.17677669529663687f;    // 32^-0.5

// ---------- kernel 0: weight convert fp32 -> bf16 ----------
__global__ void k_cvt(const float* __restrict__ wq, const float* __restrict__ wp,
                      unsigned short* __restrict__ wqb, unsigned short* __restrict__ wpb) {
    const int NQ = COUT3 * CDIM;       // 786432
    const int NP = CDIM * CDIM;        // 262144
    for (int i = blockIdx.x * blockDim.x + threadIdx.x; i < NQ + NP; i += gridDim.x * blockDim.x) {
        if (i < NQ) wqb[i] = f2bf(wq[i]);
        else        wpb[i - NQ] = f2bf(wp[i - NQ]);
    }
}

// ---------- kernel 0b: fused (rel-pos bias + mask) table, padded 64x64, -1e30 pad ----------
__global__ void k_bias(const float* __restrict__ table, const float* __restrict__ mask,
                       const int* __restrict__ rel, float* __restrict__ bmout) {
    const int h = blockIdx.x;                     // 16 heads
    for (int idx = threadIdx.x; idx < 64 * 64; idx += blockDim.x) {
        int m = idx >> 6, kc = idx & 63;
        float v = -1e30f;                         // pad keys -> softmax prob exactly 0
        if (m < NTOK && kc < NTOK)
            v = table[rel[m * NTOK + kc] * HEADS + h] + mask[m * NTOK + kc];
        bmout[h * 4096 + idx] = v;
    }
}

// ---------- kernel 0c: zero key-pad of transposed V buffer ----------
__global__ void k_vpad(unsigned short* __restrict__ vt) {
    const long long total = (long long)BWIN * HEADS * HD * (64 - NTOK);   // 15.7M halves
    for (long long i = (long long)blockIdx.x * blockDim.x + threadIdx.x; i < total;
         i += (long long)gridDim.x * blockDim.x) {
        int t  = (int)(i / (HD * (64 - NTOK)));
        int rr = (int)(i % (HD * (64 - NTOK)));
        int d  = rr / (64 - NTOK);
        int n  = NTOK + rr % (64 - NTOK);
        vt[(size_t)t * (HD * 64) + d * 64 + n] = 0;
    }
}

// ---------- kernel 1: QKV GEMM  [100352,512] x [512,1536] ----------
// block = 256 thr (8 waves), one 16-row token tile; A staged bf16 in LDS (stride 520).
// Each wave owns 12 N-tiles (96 accum VGPRs), K-outer loop -> 192 WMMAs/wave.
// Epilogue scatters bf16 into q (scaled), k, and TRANSPOSED v layouts.
__global__ __launch_bounds__(256) void k_qkv_gemm(
    const float* __restrict__ x, const float* __restrict__ b_qkv,
    const unsigned short* __restrict__ wq,
    unsigned short* __restrict__ qo, unsigned short* __restrict__ ko,
    unsigned short* __restrict__ vo) {
    __shared__ unsigned short As[16 * 520];
    const int row0 = blockIdx.x * 16;

    // cooperative load x tile (fp32) -> bf16 LDS
    for (int i = 0; i < 8; ++i) {
        int idx = i * 256 + threadIdx.x;           // 2048 float4 groups
        int r = idx >> 7;
        int c = (idx & 127) << 2;
        float4 f = *reinterpret_cast<const float4*>(x + (size_t)(row0 + r) * CDIM + c);
        uint2 p;
        p.x = (unsigned)f2bf(f.x) | ((unsigned)f2bf(f.y) << 16);
        p.y = (unsigned)f2bf(f.z) | ((unsigned)f2bf(f.w) << 16);
        *reinterpret_cast<uint2*>(&As[r * 520 + c]) = p;
    }
    __syncthreads();

    const int lane = threadIdx.x & 31;
    const int wv = threadIdx.x >> 5;
    const int nl = lane & 15, hi = lane >> 4;

    const v8f vz = {0.f, 0.f, 0.f, 0.f, 0.f, 0.f, 0.f, 0.f};
    v8f acc[12];
#pragma unroll
    for (int j = 0; j < 12; ++j) acc[j] = vz;

    for (int kb = 0; kb < 16; ++kb) {              // K = 512 in steps of 32
        const unsigned short* a0 = &As[nl * 520 + kb * 32 + hi * 8];
        v16bf a = load_frag(a0, a0 + 16);
#pragma unroll
        for (int j = 0; j < 12; ++j) {
            int col = (wv * 12 + j) * 16 + nl;
            const unsigned short* bp = wq + (size_t)col * CDIM + kb * 32 + hi * 8;
            acc[j] = wmma_bf16(a, load_frag(bp, bp + 16), acc[j]);
        }
    }

#pragma unroll
    for (int j = 0; j < 12; ++j) {
        int o = (wv * 12 + j) * 16 + nl;
        int which = o / CDIM;                      // tile-uniform (512 % 16 == 0)
        int rem = o - which * CDIM;
        int head = rem >> 5, d = rem & 31;
        float bias = b_qkv[o];
#pragma unroll
        for (int r = 0; r < 8; ++r) {
            int m = row0 + r + hi * 8;             // global token
            int bw = m / NTOK;
            int n = m - bw * NTOK;
            float val = acc[j][r] + bias;
            size_t bh = (size_t)bw * HEADS + head;
            if (which == 0)      qo[(bh * NTOK + n) * HD + d] = f2bf(val * SCALE);
            else if (which == 1) ko[(bh * NTOK + n) * HD + d] = f2bf(val);
            else                 vo[(bh * HD + d) * 64 + n]   = f2bf(val);  // transposed + padded
        }
    }
}

// ---------- kernel 2: attention, one wave per (window, head) ----------
__global__ __launch_bounds__(128) void k_attn(
    const unsigned short* __restrict__ qg, const unsigned short* __restrict__ kg,
    const unsigned short* __restrict__ vtg, const float* __restrict__ biasmask,
    unsigned short* __restrict__ ctx) {
    __shared__ unsigned short P[4][64 * 72];       // per-wave P tile, bf16, stride 72
    const int lane = threadIdx.x & 31;
    const int wv = threadIdx.x >> 5;
    const int task = blockIdx.x * 4 + wv;          // 32768 tasks
    const int bw = task >> 4, h = task & 15;
    const unsigned short* q  = qg  + (size_t)task * NTOK * HD;
    const unsigned short* k  = kg  + (size_t)task * NTOK * HD;
    const unsigned short* vt = vtg + (size_t)task * HD * 64;
    const float* bm = biasmask + h * 4096;
    unsigned short* Pw = P[wv];
    const int nl = lane & 15, hi = lane >> 4;
    const v8f vz = {0.f, 0.f, 0.f, 0.f, 0.f, 0.f, 0.f, 0.f};

    // phase 1: S = q k^T (+bias+mask), softmax, P -> LDS bf16
    for (int mt = 0; mt < 4; ++mt) {
        const unsigned short* a0 = q + (mt * 16 + nl) * HD + hi * 8;
        v16bf aq = load_frag(a0, a0 + 16);
        v8f s[4];
#pragma unroll
        for (int nt = 0; nt < 4; ++nt) {
            const unsigned short* bp = k + (nt * 16 + nl) * HD + hi * 8;
            s[nt] = wmma_bf16(aq, load_frag(bp, bp + 16), vz);   // K=32=hd, one WMMA
        }
#pragma unroll
        for (int nt = 0; nt < 4; ++nt)
#pragma unroll
            for (int r = 0; r < 8; ++r)
                s[nt][r] += bm[(mt * 16 + r + hi * 8) * 64 + nt * 16 + nl];

#pragma unroll
        for (int r = 0; r < 8; ++r) {
            float mx = fmaxf(fmaxf(s[0][r], s[1][r]), fmaxf(s[2][r], s[3][r]));
            mx = fmaxf(mx, __shfl_xor(mx, 1, 32));
            mx = fmaxf(mx, __shfl_xor(mx, 2, 32));
            mx = fmaxf(mx, __shfl_xor(mx, 4, 32));
            mx = fmaxf(mx, __shfl_xor(mx, 8, 32));   // masks<=8 stay inside 16-lane half = row
            float e0 = __expf(s[0][r] - mx);
            float e1 = __expf(s[1][r] - mx);
            float e2 = __expf(s[2][r] - mx);
            float e3 = __expf(s[3][r] - mx);
            float sm = e0 + e1 + e2 + e3;
            sm += __shfl_xor(sm, 1, 32);
            sm += __shfl_xor(sm, 2, 32);
            sm += __shfl_xor(sm, 4, 32);
            sm += __shfl_xor(sm, 8, 32);
            float inv = 1.0f / sm;
            int m = mt * 16 + r + hi * 8;
            Pw[m * 72 +      nl] = f2bf(e0 * inv);
            Pw[m * 72 + 16 + nl] = f2bf(e1 * inv);
            Pw[m * 72 + 32 + nl] = f2bf(e2 * inv);
            Pw[m * 72 + 48 + nl] = f2bf(e3 * inv);
        }
    }
    // same-wave DS ordering guarantees stores visible to following loads

    // phase 2: O = P(16x64) @ Vt
#pragma unroll
    for (int mt = 0; mt < 4; ++mt) {
#pragma unroll
        for (int dt = 0; dt < 2; ++dt) {
            v8f o = vz;
#pragma unroll
            for (int kt = 0; kt < 2; ++kt) {
                const unsigned short* pa = &Pw[(mt * 16 + nl) * 72 + kt * 32 + hi * 8];
                v16bf ap = load_frag(pa, pa + 16);
                const unsigned short* pb = vt + (dt * 16 + nl) * 64 + kt * 32 + hi * 8;
                o = wmma_bf16(ap, load_frag(pb, pb + 16), o);
            }
#pragma unroll
            for (int r = 0; r < 8; ++r) {
                int m = mt * 16 + r + hi * 8;
                if (m < NTOK)
                    ctx[((size_t)bw * NTOK + m) * CDIM + h * HD + dt * 16 + nl] = f2bf(o[r]);
            }
        }
    }
}

// ---------- kernel 3: proj GEMM [100352,512] x [512,512] -> fp32 out ----------
__global__ __launch_bounds__(256) void k_proj(
    const unsigned short* __restrict__ ctx, const unsigned short* __restrict__ wp,
    const float* __restrict__ b_proj, float* __restrict__ out) {
    __shared__ unsigned short As[16 * 520];
    const int row0 = blockIdx.x * 16;
    for (int i = 0; i < 4; ++i) {
        int idx = i * 256 + threadIdx.x;           // 1024 chunks of 8 halves
        int r = idx >> 6;
        int c = (idx & 63) << 3;
        *reinterpret_cast<uint4*>(&As[r * 520 + c]) =
            *reinterpret_cast<const uint4*>(ctx + (size_t)(row0 + r) * CDIM + c);
    }
    __syncthreads();

    const int lane = threadIdx.x & 31;
    const int wv = threadIdx.x >> 5;
    const int nl = lane & 15, hi = lane >> 4;
    const v8f vz = {0.f, 0.f, 0.f, 0.f, 0.f, 0.f, 0.f, 0.f};
    v8f acc[4];
#pragma unroll
    for (int j = 0; j < 4; ++j) acc[j] = vz;

    for (int kb = 0; kb < 16; ++kb) {
        const unsigned short* a0 = &As[nl * 520 + kb * 32 + hi * 8];
        v16bf a = load_frag(a0, a0 + 16);
#pragma unroll
        for (int j = 0; j < 4; ++j) {
            int col = (wv * 4 + j) * 16 + nl;
            const unsigned short* bp = wp + (size_t)col * CDIM + kb * 32 + hi * 8;
            acc[j] = wmma_bf16(a, load_frag(bp, bp + 16), acc[j]);
        }
    }
#pragma unroll
    for (int j = 0; j < 4; ++j) {
        int o = (wv * 4 + j) * 16 + nl;
        float bias = b_proj[o];
#pragma unroll
        for (int r = 0; r < 8; ++r)
            out[(size_t)(row0 + r + hi * 8) * CDIM + o] = acc[j][r] + bias;
    }
}

// ---------- launch ----------
extern "C" void kernel_launch(void* const* d_in, const int* in_sizes, int n_in,
                              void* d_out, int out_size, void* d_ws, size_t ws_size,
                              hipStream_t stream) {
    const float* x          = (const float*)d_in[0];
    const float* mask       = (const float*)d_in[1];
    const float* w_qkv      = (const float*)d_in[2];
    const float* b_qkv      = (const float*)d_in[3];
    const float* bias_table = (const float*)d_in[4];
    const float* w_proj     = (const float*)d_in[5];
    const float* b_proj     = (const float*)d_in[6];
    const int*   rel_index  = (const int*)d_in[7];

    char* ws = (char*)d_ws;
    // 256B-aligned workspace layout (total ~425 MB)
    unsigned short* wqb  = (unsigned short*)(ws + 0);           //  1.5 MB  w_qkv bf16
    unsigned short* wpb  = (unsigned short*)(ws + 1572864);     //  0.5 MB  w_proj bf16
    float*          bmsk = (float*)        (ws + 2097152);      //  256 KB  bias+mask [16][64][64]
    unsigned short* qb   = (unsigned short*)(ws + 2359296);     //  98 MB   q  [b,h,49,32] (scaled)
    unsigned short* kb   = (unsigned short*)(ws + 105119744);   //  98 MB   k  [b,h,49,32]
    unsigned short* vtb  = (unsigned short*)(ws + 207880192);   // 128 MB   vT [b,h,32,64]
    unsigned short* ctx  = (unsigned short*)(ws + 342097920);   //  98 MB   ctx [b,49,512]

    k_cvt <<<512,  256, 0, stream>>>(w_qkv, w_proj, wqb, wpb);
    k_bias<<<16,   256, 0, stream>>>(bias_table, mask, rel_index, bmsk);
    k_vpad<<<8192, 256, 0, stream>>>(vtb);
    k_qkv_gemm<<<MROWS / 16, 256, 0, stream>>>(x, b_qkv, wqb, qb, kb, vtb);
    k_attn<<<(BWIN * HEADS) / 4, 128, 0, stream>>>(qb, kb, vtb, bmsk, ctx);
    k_proj<<<MROWS / 16, 256, 0, stream>>>(ctx, wpb, b_proj, (float*)d_out);
}
